// DRSformerRefFusion_89146341196102
// MI455X (gfx1250) — matrix operand order repaired and checked by reading, not compiled, not run on hardware
//
#include <hip/hip_runtime.h>
#include <hip/hip_bf16.h>
#include <math.h>

// ---------------------------------------------------------------------------
// DRSformer attention block for MI455X (gfx1250, wave32, WMMA).
//
//   xT_bf   = transpose+cvt(x)              (tiled LDS transpose, fp32->bf16)
//   qkv_pre = Wqkv * x                      (WMMA GEMM, M=576 K=192 N=16384)
//   qkv_dw  = dwconv3x3(qkv_pre)            (fp32, memory bound)
//   qk_bf   = L2-normalize(q,k) -> bf16     (row norms over 16384 tokens)
//   vT_bf   = transpose+cvt(v)
//   attn    = q * k^T                       (WMMA GEMM, split-K=32, partials)
//   Acomb   = sum_i w_i*softmax(topk_i(attn*temp))   (768 tiny rows)
//   avout   = Acomb * v                     (WMMA GEMM, M=48 N=16384 K=48)
//   avT_bf  = transpose+cvt(avout)
//   out     = Wout * avout                  (WMMA GEMM, M=192 K=192 N=16384)
//
// Full tiles stage A/B via GLOBAL_LOAD_ASYNC_TO_LDS_B128 into double-buffered
// LDS (ASYNCcnt), overlapping the next tile's DMA with the current WMMAs.
// Edge tiles use a masked vector-register path with zero-fill.
// ---------------------------------------------------------------------------

typedef __attribute__((ext_vector_type(16))) __bf16 v16bf;
typedef __attribute__((ext_vector_type(8)))  __bf16 v8bf;
typedef __attribute__((ext_vector_type(8)))  float  v8f;
typedef __attribute__((ext_vector_type(4)))  unsigned int u32x4;
typedef __attribute__((__vector_size__(4 * sizeof(int)))) int v4i;  // builtin's pointee type

union V16U { v16bf v; v8bf h[2]; };

#define TILE_M 64
#define TILE_N 128
#define KSTEP  32
#define LDS_STRIDE 40   // padded halfwords -> conflict-free 16B lane reads

#if defined(__gfx1250__) && __has_builtin(__builtin_amdgcn_global_load_async_to_lds_b128)
#define USE_ASYNC_LDS 1
#else
#define USE_ASYNC_LDS 0
#endif

__device__ __forceinline__ void async_wait0() {
#if USE_ASYNC_LDS
#if __has_builtin(__builtin_amdgcn_s_wait_asynccnt)
    __builtin_amdgcn_s_wait_asynccnt(0);
#else
    asm volatile("s_wait_asynccnt 0x0" ::: "memory");
#endif
#endif
}

// ------------------------- generic WMMA GEMM -------------------------------
// C(M,Nreal) = A(M,K) * B(K,Nreal); A addressed [m*lda + k] (row-major),
// B addressed [n*ldb + k] (k-contiguous per column). bf16 in, f32 out.
// z decomposes into (batch=z/nh, head=z%nh). blockIdx.x = nTile + nTiles*kc;
// with kSplit>1 each chunk kc writes its partial C at + kc*sCk (deterministic).
__global__ __launch_bounds__(256) void gemm_wmma_bf16(
    const __bf16* __restrict__ A, int lda, long long sAb, long long sAh,
    const __bf16* __restrict__ B, int ldb, long long sBb, long long sBh,
    float* __restrict__ C, int ldc, long long sCb, long long sCh,
    int M, int Nreal, int K, int nh, int nTiles, int kSplit, long long sCk)
{
    __shared__ alignas(16) __bf16 Alds[2][TILE_M * LDS_STRIDE];
    __shared__ alignas(16) __bf16 Blds[2][TILE_N * LDS_STRIDE];

    const int z  = blockIdx.z;
    const int bb = z / nh, hh = z % nh;
    const __bf16* Az = A + bb * sAb + hh * sAh;
    const __bf16* Bz = B + bb * sBb + hh * sBh;

    const int nt = blockIdx.x % nTiles;
    const int kc = blockIdx.x / nTiles;
    const int kPer = K / kSplit;
    const int kBeg = kc * kPer, kEnd = kBeg + kPer;
    float* Cz = C + bb * sCb + hh * sCh + kc * sCk;

    const int n0 = nt * TILE_N;
    const int m0 = blockIdx.y * TILE_M;

    const int tid  = threadIdx.x;
    const int lane = tid & 31;
    const int wave = tid >> 5;        // 8 wave32s
    const int wm   = wave & 3;        // m strip (16 rows)
    const int wn   = wave >> 2;       // n half (64 cols)
    const int lm   = lane & 15;
    const int lh   = lane >> 4;

    v8f acc[4];
    #pragma unroll
    for (int s = 0; s < 4; ++s) acc[s] = (v8f){};

    // staging coordinates: A 64 rows x 4 chunks of 8; B 128 cols x 2 chunks of 16
    const int arow = tid >> 2, achk = (tid & 3) << 3;
    const int bn   = tid & 127, bkh = (tid >> 7) << 4;     // 0 or 16

    const bool aRowOk = (m0 + arow) < M;
    const bool bColOk = (n0 + bn)   < Nreal;
    const bool fullMN = (m0 + TILE_M <= M) && (n0 + TILE_N <= Nreal);
    const __bf16* aSrc = Az + (long long)(m0 + arow) * lda + achk;
    const __bf16* bSrc = Bz + (long long)(n0 + bn)   * ldb + bkh;
    const int aOff = arow * LDS_STRIDE + achk;
    const int bOff = bn   * LDS_STRIDE + bkh;

    auto stage = [&](int kk, int p) {
        if (fullMN && (kk + KSTEP) <= K) {
#if USE_ASYNC_LDS
            // DMA straight to LDS; tracked by ASYNCcnt, no VGPR round-trip.
            __builtin_amdgcn_global_load_async_to_lds_b128(
                (v4i*)(aSrc + kk), (v4i*)&Alds[p][aOff], 0, 0);
            __builtin_amdgcn_global_load_async_to_lds_b128(
                (v4i*)(bSrc + kk), (v4i*)&Blds[p][bOff], 0, 0);
            __builtin_amdgcn_global_load_async_to_lds_b128(
                (v4i*)(bSrc + kk + 8), (v4i*)&Blds[p][bOff + 8], 0, 0);
#else
            u32x4 av = *reinterpret_cast<const u32x4*>(aSrc + kk);
            u32x4 b0 = *reinterpret_cast<const u32x4*>(bSrc + kk);
            u32x4 b1 = *reinterpret_cast<const u32x4*>(bSrc + kk + 8);
            *reinterpret_cast<u32x4*>(&Alds[p][aOff])     = av;
            *reinterpret_cast<u32x4*>(&Blds[p][bOff])     = b0;
            *reinterpret_cast<u32x4*>(&Blds[p][bOff + 8]) = b1;
#endif
        } else {
            u32x4 av = (u32x4){0u, 0u, 0u, 0u};
            u32x4 b0 = (u32x4){0u, 0u, 0u, 0u};
            u32x4 b1 = (u32x4){0u, 0u, 0u, 0u};
            if (aRowOk && (kk + achk + 8) <= K)
                av = *reinterpret_cast<const u32x4*>(aSrc + kk);
            if (bColOk && (kk + bkh + 16) <= K) {
                b0 = *reinterpret_cast<const u32x4*>(bSrc + kk);
                b1 = *reinterpret_cast<const u32x4*>(bSrc + kk + 8);
            }
            *reinterpret_cast<u32x4*>(&Alds[p][aOff])     = av;
            *reinterpret_cast<u32x4*>(&Blds[p][bOff])     = b0;
            *reinterpret_cast<u32x4*>(&Blds[p][bOff + 8]) = b1;
        }
    };

    // prologue: fill buffer 0
    stage(kBeg, 0);
    async_wait0();
    __syncthreads();

    int p = 0;
    for (int kk = kBeg; kk < kEnd; kk += KSTEP) {
        // stream next tile into the other buffer while computing this one
        if (kk + KSTEP < kEnd) stage(kk + KSTEP, p ^ 1);

        // A fragment: lane<16 -> K{0..7,16..23}, lane>=16 -> K{8..15,24..31}
        V16U af;
        {
            const int r  = wm * 16 + lm;
            const int k0 = lh * 8;
            af.h[0] = *reinterpret_cast<const v8bf*>(&Alds[p][r * LDS_STRIDE + k0]);
            af.h[1] = *reinterpret_cast<const v8bf*>(&Alds[p][r * LDS_STRIDE + k0 + 16]);
        }
        #pragma unroll
        for (int s = 0; s < 4; ++s) {
            const int nloc = wn * 64 + s * 16 + lm;   // lane column; lh picks K{0..7,16..23} or K{8..15,24..31}
            V16U bfv;
            bfv.h[0] = *reinterpret_cast<const v8bf*>(&Blds[p][nloc * LDS_STRIDE + lh * 8]);
            bfv.h[1] = *reinterpret_cast<const v8bf*>(&Blds[p][nloc * LDS_STRIDE + lh * 8 + 16]);
            acc[s] = __builtin_amdgcn_wmma_f32_16x16x32_bf16(
                false, af.v, false, bfv.v, (short)0, acc[s], false, false);
        }

        async_wait0();          // next tile's DMA landed in LDS
        __syncthreads();
        p ^= 1;
    }

    // epilogue: C/D layout VGPR r -> M = r + 8*(lane>=16), N = lane%16
    #pragma unroll
    for (int s = 0; s < 4; ++s) {
        const int gcol = n0 + wn * 64 + s * 16 + lm;
        #pragma unroll
        for (int r = 0; r < 8; ++r) {
            const int grow = m0 + wm * 16 + lh * 8 + r;
            if (grow < M && gcol < Nreal)
                Cz[(long long)grow * ldc + gcol] = acc[s][r];
        }
    }
}

// ---------------- tiled transpose + fp32 -> bf16 ---------------------------
__global__ __launch_bounds__(256) void transpose_cvt_k(
    const float* __restrict__ in, long long sInz, int ldIn,
    __bf16* __restrict__ out, long long sOutz, int ldOut)
{
    __shared__ float t[32][33];
    const int n0 = blockIdx.x * 32, r0 = blockIdx.y * 32;
    const float* iz = in + blockIdx.z * sInz;
    __bf16*      oz = out + blockIdx.z * sOutz;
    const int tx = threadIdx.x & 31, ty = threadIdx.x >> 5;   // 32 x 8
    #pragma unroll
    for (int j = 0; j < 4; ++j)
        t[ty + j * 8][tx] = iz[(long long)(r0 + ty + j * 8) * ldIn + n0 + tx];
    __syncthreads();
    #pragma unroll
    for (int j = 0; j < 4; ++j)
        oz[(long long)(n0 + ty + j * 8) * ldOut + r0 + tx] = (__bf16)t[tx][ty + j * 8];
}

// ------------------------- fp32 -> bf16 convert ----------------------------
__global__ void f32_to_bf16_k(const float* __restrict__ in,
                              __bf16* __restrict__ out, long long n)
{
    long long i = (long long)blockIdx.x * blockDim.x + threadIdx.x;
    if (i < n) out[i] = (__bf16)in[i];
}

// -------------------- split-K partial reduction ----------------------------
__global__ void reduce_splitk_k(const float* __restrict__ part,
                                float* __restrict__ out, long long n, int chunks)
{
    long long i = (long long)blockIdx.x * blockDim.x + threadIdx.x;
    if (i >= n) return;
    float s = 0.f;
    for (int c = 0; c < chunks; ++c) s += part[c * n + i];
    out[i] = s;
}

// ------------------------- depthwise 3x3, pad 1 ----------------------------
__global__ void dwconv3x3_k(const float* __restrict__ in,
                            const float* __restrict__ w,
                            float* __restrict__ out,
                            int Cn, int H, int W, long long total)
{
    long long i = (long long)blockIdx.x * blockDim.x + threadIdx.x;
    if (i >= total) return;
    int x  = (int)(i % W);
    int y  = (int)((i / W) % H);
    int ch = (int)((i / ((long long)W * H)) % Cn);
    long long plane = i - x - (long long)y * W;
    const float* wp = w + ch * 9;
    float s = 0.f;
    #pragma unroll
    for (int dy = -1; dy <= 1; ++dy) {
        int yy = y + dy;
        if (yy < 0 || yy >= H) continue;
        #pragma unroll
        for (int dx = -1; dx <= 1; ++dx) {
            int xx = x + dx;
            if (xx < 0 || xx >= W) continue;
            s += in[plane + (long long)yy * W + xx] * wp[(dy + 1) * 3 + (dx + 1)];
        }
    }
    out[i] = s;
}

// -------- L2-normalize q,k rows over tokens; convert to bf16 ---------------
__global__ __launch_bounds__(256) void norm_convert_k(
    const float* __restrict__ qkv, __bf16* __restrict__ out, int N)
{
    __shared__ float red[256];
    const int row = blockIdx.x;                 // over B*384
    const int b   = row / 384, ch = row % 384;
    const long long ibase = ((long long)b * 576 + ch) * N;
    const long long obase = (long long)row * N;
    const int t = threadIdx.x;

    float ss = 0.f;
    for (int i = t; i < N; i += 256) {
        float v = qkv[ibase + i];
        ss += v * v;
    }
    red[t] = ss;
    __syncthreads();
    for (int off = 128; off > 0; off >>= 1) {
        if (t < off) red[t] += red[t + off];
        __syncthreads();
    }
    const float scale = 1.0f / fmaxf(sqrtf(red[0]), 1e-12f);
    for (int i = t; i < N; i += 256)
        out[obase + i] = (__bf16)(qkv[ibase + i] * scale);
}

// ------- combined top-k masked softmax: Acomb = sum_i w_i*softmax_i --------
__global__ __launch_bounds__(64) void topk_softmax_combine_k(
    const float* __restrict__ attn, const float* __restrict__ temp,
    const float* __restrict__ attnw, __bf16* __restrict__ acomb,
    int C, int nh)
{
    __shared__ float a[64], e[64];
    __shared__ int   rk[64];
    const int row = blockIdx.x;            // over B*nh*C
    const int h   = (row / C) % nh;
    const long long base = (long long)row * C;
    const int t = threadIdx.x;

    if (t < C) a[t] = attn[base + t] * temp[h];
    __syncthreads();

    if (t < C) {
        float ai = a[t];
        float mx = -INFINITY;
        int cnt = 0;
        for (int j = 0; j < C; ++j) {
            float aj = a[j];
            mx = fmaxf(mx, aj);
            if (aj > ai || (aj == ai && j < t)) ++cnt;   // topk tie-break by index
        }
        rk[t] = cnt;
        e[t] = __expf(ai - mx);   // top-1 is in every kept set -> mx valid
    }
    __syncthreads();

    if (t < C) {
        const int kv[4] = { C / 2, C * 2 / 3, C * 3 / 4, C * 4 / 5 };
        float den[4] = {0.f, 0.f, 0.f, 0.f};
        for (int j = 0; j < C; ++j) {
            float ej = e[j]; int rj = rk[j];
            #pragma unroll
            for (int i = 0; i < 4; ++i) if (rj < kv[i]) den[i] += ej;
        }
        float o = 0.f;
        #pragma unroll
        for (int i = 0; i < 4; ++i)
            if (rk[t] < kv[i]) o += attnw[i] * (e[t] / den[i]);
        acomb[base + t] = (__bf16)o;
    }
}

// ---------------------------------------------------------------------------
extern "C" void kernel_launch(void* const* d_in, const int* in_sizes, int n_in,
                              void* d_out, int out_size, void* d_ws, size_t ws_size,
                              hipStream_t stream)
{
    const int  Bb = 4, DIM = 192, TRI = 576, HEADS = 4, Cc = 48, H = 128, W = 128;
    const int  KSPLIT = 32;                                // attn split-K chunks
    const long long N = (long long)H * W;                  // 16384 tokens

    const float* x     = (const float*)d_in[0];
    const float* w_qkv = (const float*)d_in[1];
    const float* w_dw  = (const float*)d_in[2];
    const float* w_out = (const float*)d_in[3];
    const float* temp  = (const float*)d_in[4];
    const float* attnw = (const float*)d_in[5];

    // workspace carve-out
    char* ws = (char*)d_ws;
    size_t off = 0;
    auto carve = [&](size_t bytes) { char* p = ws + off; off = (off + bytes + 255) & ~(size_t)255; return p; };
    __bf16* xT_bf     = (__bf16*)carve((size_t)Bb * DIM * N * 2);   // [b][n][192]
    __bf16* wqkv_bf   = (__bf16*)carve((size_t)TRI * DIM * 2);
    __bf16* wout_bf   = (__bf16*)carve((size_t)DIM * DIM * 2);
    float*  qkv_pre   = (float*) carve((size_t)Bb * TRI * N * 4);
    float*  qkv_dw    = (float*) carve((size_t)Bb * TRI * N * 4);
    __bf16* qk_bf     = (__bf16*)carve((size_t)Bb * 2 * DIM * N * 2); // [b][384][N]
    __bf16* vT_bf     = (__bf16*)carve((size_t)Bb * DIM * N * 2);   // [b][n][192]
    float*  attn      = (float*) carve((size_t)Bb * HEADS * Cc * Cc * 4);
    float*  attn_part = (float*) carve((size_t)KSPLIT * Bb * HEADS * Cc * Cc * 4);
    __bf16* acomb_bf  = (__bf16*)carve((size_t)Bb * HEADS * Cc * Cc * 2);
    float*  avout     = (float*) carve((size_t)Bb * DIM * N * 4);   // [b][192][N]
    __bf16* avT_bf    = (__bf16*)carve((size_t)Bb * DIM * N * 2);   // [b][n][192]

    const long long attnElems = (long long)Bb * HEADS * Cc * Cc;    // 36864

    // 1) weights -> bf16; x -> transposed bf16 (xT[b][n][c])
    {
        long long n = (long long)TRI * DIM;
        f32_to_bf16_k<<<(unsigned)((n + 255) / 256), 256, 0, stream>>>(w_qkv, wqkv_bf, n);
        n = (long long)DIM * DIM;
        f32_to_bf16_k<<<(unsigned)((n + 255) / 256), 256, 0, stream>>>(w_out, wout_bf, n);
        dim3 g((unsigned)(N / 32), DIM / 32, Bb);
        transpose_cvt_k<<<g, 256, 0, stream>>>(x, DIM * N, (int)N, xT_bf, N * DIM, DIM);
    }

    // 2) qkv_pre = Wqkv(576x192) * x(192xN) per batch  (async-LDS fast path)
    {
        dim3 g((unsigned)(N / TILE_N), TRI / TILE_M, Bb);
        gemm_wmma_bf16<<<g, 256, 0, stream>>>(
            wqkv_bf, DIM, 0, 0,
            xT_bf, DIM, N * DIM, 0,
            qkv_pre, (int)N, (long long)TRI * N, 0,
            TRI, (int)N, DIM, 1, (int)(N / TILE_N), 1, 0);
    }

    // 3) depthwise 3x3
    {
        long long total = (long long)Bb * TRI * N;
        dwconv3x3_k<<<(unsigned)((total + 255) / 256), 256, 0, stream>>>(
            qkv_pre, w_dw, qkv_dw, TRI, H, W, total);
    }

    // 4a) normalize q,k rows -> qk_bf [b][384][N]
    norm_convert_k<<<Bb * 2 * DIM, 256, 0, stream>>>(qkv_dw, qk_bf, (int)N);
    // 4b) v -> transposed bf16 vT[b][n][192]
    {
        dim3 g((unsigned)(N / 32), DIM / 32, Bb);
        transpose_cvt_k<<<g, 256, 0, stream>>>(qkv_dw + 2 * DIM * N, TRI * N, (int)N,
                                               vT_bf, N * DIM, DIM);
    }

    // 5) attn = q * k^T  (48x48, K=16384) split-K=32 -> deterministic partials
    {
        dim3 g(KSPLIT, 1, Bb * HEADS);
        gemm_wmma_bf16<<<g, 256, 0, stream>>>(
            qk_bf,           (int)N, 2 * DIM * N, Cc * N,     // q rows
            qk_bf + DIM * N, (int)N, 2 * DIM * N, Cc * N,     // k cols [d][token]
            attn_part, Cc, (long long)HEADS * Cc * Cc, (long long)Cc * Cc,
            Cc, Cc, (int)N, HEADS, 1, KSPLIT, attnElems);
        reduce_splitk_k<<<(unsigned)((attnElems + 255) / 256), 256, 0, stream>>>(
            attn_part, attn, attnElems, KSPLIT);
    }

    // 6) combined top-k softmax (4 k-values folded into one matrix)
    topk_softmax_combine_k<<<Bb * HEADS * Cc, 64, 0, stream>>>(
        attn, temp, attnw, acomb_bf, Cc, HEADS);

    // 7) avout = Acomb(48x48) * v(48xN)   (B = vT[b][n][192] + h*48)
    {
        dim3 g((unsigned)(N / TILE_N), 1, Bb * HEADS);
        gemm_wmma_bf16<<<g, 256, 0, stream>>>(
            acomb_bf, Cc, (long long)HEADS * Cc * Cc, (long long)Cc * Cc,
            vT_bf, DIM, N * DIM, Cc,
            avout, (int)N, (long long)DIM * N, (long long)Cc * N,
            Cc, (int)N, Cc, HEADS, (int)(N / TILE_N), 1, 0);
    }

    // 8) avout -> transposed bf16 avT[b][n][192]
    {
        dim3 g((unsigned)(N / 32), DIM / 32, Bb);
        transpose_cvt_k<<<g, 256, 0, stream>>>(avout, DIM * N, (int)N,
                                               avT_bf, N * DIM, DIM);
    }

    // 9) out = Wout(192x192) * avout(192xN) per batch -> d_out (fp32)
    {
        dim3 g((unsigned)(N / TILE_N), DIM / TILE_M, Bb);
        gemm_wmma_bf16<<<g, 256, 0, stream>>>(
            wout_bf, DIM, 0, 0,
            avT_bf, DIM, N * DIM, 0,
            (float*)d_out, (int)N, (long long)DIM * N, 0,
            DIM, (int)N, DIM, 1, (int)(N / TILE_N), 1, 0);
    }
    (void)in_sizes; (void)n_in; (void)out_size; (void)ws_size;
}
